// DynamicVoxelization_21526376087724
// MI455X (gfx1250) — matrix operand order repaired and checked by reading, not compiled, not run on hardware
//
#include <hip/hip_runtime.h>

typedef __attribute__((ext_vector_type(16))) _Float16 v16h;
typedef __attribute__((ext_vector_type(8)))  float    v8f;

#define NPERB   100000          // points per batch
#define NPTS    400000          // B*N
#define NGROUPS (NPTS / 16)     // 25000 wave-groups of 16 points
#define MLPD    128
#define GRIDXY  468
#define VOLV    (468 * 468 * 1) // 219024 voxels per batch

// ---------------------------------------------------------------------------
// Zero-fill the output grid with wide B128 stores.  0u bit-pattern == 0.0f.
// ---------------------------------------------------------------------------
__global__ void fill_zero_b128(float4* __restrict__ p, int n4) {
    int i      = blockIdx.x * blockDim.x + threadIdx.x;
    int stride = gridDim.x * blockDim.x;
    const float4 z = make_float4(0.f, 0.f, 0.f, 0.f);
    for (; i < n4; i += stride) p[i] = z;
}

// ---------------------------------------------------------------------------
// Fused voxelize + MLP (WMMA f16) + atomic-max scatter.
// One wave handles 16 points per iteration:
//   A (16x32 f16)  = [feat(16) | delta(3) | 0-pad] rows, mask-zeroed
//   B (32x16 f16)  = W columns (8 tiles of N=16), hoisted to registers
//   C (16x16 f32)  = 8 x v_wmma_f32_16x16x32_f16, then +bias, ReLU,
//                    predicated GLOBAL_ATOMIC_MAX_U32 scatter.
// ---------------------------------------------------------------------------
__global__ __launch_bounds__(256)
void voxel_mlp_scatter(const float* __restrict__ pxyz,
                       const float* __restrict__ pfeat,
                       const unsigned char* __restrict__ pmask,
                       const float* __restrict__ W,
                       const float* __restrict__ bias,
                       unsigned int* __restrict__ out)
{
    const int lane  = threadIdx.x & 31;
    const int wave  = threadIdx.x >> 5;
    const int m     = lane & 15;   // point-in-group (row M)
    const int half  = lane >> 4;   // K-half owned by this lane
    const int wavesPerBlock = blockDim.x >> 5;
    const int waveId = blockIdx.x * wavesPerBlock + wave;
    const int nWaves = gridDim.x * wavesPerBlock;

    // ---- hoist B fragments (W is 19x128 row-major) and bias into registers
    // 16-bit B 32x16 layout: lanes 0-15 hold K=0..7,16..23 of column N=lane,
    // lanes 16-31 hold K=8..15,24..31 of column N=lane-16.  K>=19 is zero pad.
    v16h  bfrag[8];
    float bcol[8];
    const int kbase = half ? 8 : 0;
#pragma unroll
    for (int t = 0; t < 8; ++t) {
        const int c = t * 16 + m;
#pragma unroll
        for (int e = 0; e < 16; ++e) {
            const int k = kbase + e + (e >= 8 ? 8 : 0);
            const float w = (k < 19) ? W[k * MLPD + c] : 0.0f;
            bfrag[t][e] = (_Float16)w;
        }
        bcol[t] = bias[c];
    }

    for (int g = waveId; g < NGROUPS; g += nWaves) {
        const int p = g * 16 + m;

        // ---- voxelization (both K-half lanes compute; cheap VALU)
        const float x = pxyz[p * 3 + 0];
        const float y = pxyz[p * 3 + 1];
        const float z = pxyz[p * 3 + 2];
        const float fx = floorf(x / 0.32f);
        const float fy = floorf(y / 0.32f);
        const float fz = floorf(z / 6.0f);
        const int ix = (int)fx + 234;            // - ORIGIN.x
        const int iy = (int)fy + 234;
        const int iz = (int)fz + 1;
        const bool inb = (ix >= 0) & (ix < GRIDXY) & (iy >= 0) & (iy < GRIDXY) &
                         (iz >= 0) & (iz < 1);
        const bool msk   = pmask[p] != 0;
        const bool valid = inb & msk;
        const float mf   = valid ? 1.0f : 0.0f;
        const int  bidx  = p / NPERB;
        const int  vid   = valid ? (bidx * VOLV + ix * GRIDXY + iy + iz) : -1;

        // ---- A fragment: 16-bit A 16x32 layout.
        // lane<16 : e0..7 = feat[0..7],  e8..15 = K16..23 = (dx,dy,dz,0...)
        // lane>=16: e0..7 = feat[8..15], e8..15 = K24..31 = 0
        const float* fptr = pfeat + (size_t)p * 16 + (half ? 8 : 0);
        const float4 f0 = *(const float4*)(fptr);
        const float4 f1 = *(const float4*)(fptr + 4);
        const float dx = half ? 0.0f : (x - fx * 0.32f) * mf;
        const float dy = half ? 0.0f : (y - fy * 0.32f) * mf;
        const float dz = half ? 0.0f : (z - fz * 6.0f)  * mf;

        v16h a;
        a[0]  = (_Float16)(f0.x * mf);
        a[1]  = (_Float16)(f0.y * mf);
        a[2]  = (_Float16)(f0.z * mf);
        a[3]  = (_Float16)(f0.w * mf);
        a[4]  = (_Float16)(f1.x * mf);
        a[5]  = (_Float16)(f1.y * mf);
        a[6]  = (_Float16)(f1.z * mf);
        a[7]  = (_Float16)(f1.w * mf);
        a[8]  = (_Float16)dx;
        a[9]  = (_Float16)dy;
        a[10] = (_Float16)dz;
        a[11] = (_Float16)0.0f;
        a[12] = (_Float16)0.0f;
        a[13] = (_Float16)0.0f;
        a[14] = (_Float16)0.0f;
        a[15] = (_Float16)0.0f;

        // ---- broadcast per-row voxel ids (C/D layout: lane L, vgpr r ->
        // M = r + 8*(L>=16)); lanes m and m+16 hold identical vid.
        int vrow[8];
#pragma unroll
        for (int r = 0; r < 8; ++r)
            vrow[r] = __shfl(vid, r + 8 * half, 32);

        // ---- 8 N-tiles of WMMA + bias + ReLU + predicated atomic-max scatter
#pragma unroll
        for (int t = 0; t < 8; ++t) {
            v8f cacc = {};
            cacc = __builtin_amdgcn_wmma_f32_16x16x32_f16(
                false, a, false, bfrag[t], (short)0, cacc, false, false);
            const int colOff = t * 16 + m;
#pragma unroll
            for (int r = 0; r < 8; ++r) {
                float v = cacc[r] + bcol[t];
                v = v > 0.0f ? v : 0.0f;
                // grid is 0-initialized and all values are >= 0: skipping
                // v<=0 / invalid rows is exact.  Unsigned compare == float
                // compare for non-negative IEEE floats.
                if (vrow[r] >= 0 && v > 0.0f) {
                    atomicMax(out + (unsigned)vrow[r] * 128u + colOff,
                              __float_as_uint(v));
                }
            }
        }
    }
}

extern "C" void kernel_launch(void* const* d_in, const int* in_sizes, int n_in,
                              void* d_out, int out_size, void* d_ws, size_t ws_size,
                              hipStream_t stream) {
    (void)in_sizes; (void)n_in; (void)d_ws; (void)ws_size;
    const float*         pxyz  = (const float*)d_in[0];         // [B,N,3]
    const float*         pfeat = (const float*)d_in[1];         // [B,N,16]
    const unsigned char* pmask = (const unsigned char*)d_in[2]; // [B,N] bool
    const float*         W     = (const float*)d_in[3];         // [19,128]
    const float*         bias  = (const float*)d_in[4];         // [128]

    // 1) zero-init the 449 MB output grid (B128 stores)
    const int n4 = out_size / 4;
    fill_zero_b128<<<2048, 256, 0, stream>>>((float4*)d_out, n4);

    // 2) fused voxelize + WMMA MLP + atomic-max scatter
    voxel_mlp_scatter<<<1024, 256, 0, stream>>>(
        pxyz, pfeat, pmask, W, bias, (unsigned int*)d_out);
}
